// ISA_695784702209
// MI455X (gfx1250) — compile-verified, matrix operand order checked by reading, not run on hardware
//
#include <hip/hip_runtime.h>
#include <hip/hip_bf16.h>

typedef __bf16 bf16_t;
typedef __attribute__((ext_vector_type(16))) __bf16 v16bf;
typedef __attribute__((ext_vector_type(8)))  __bf16 v8bf;
typedef __attribute__((ext_vector_type(8)))  float  v8f;

#define CIN   576
#define COUT  288
#define NP    16384        // N*H*W
#define NPS   4096         // N*(H/2)*(W/2)
#define KDFM  5184         // 576*9
#define EPSBN 1e-5f

__device__ __forceinline__ float fastrcp(float x) { return __builtin_amdgcn_rcpf(x); }
__device__ __forceinline__ float siluf(float x) { return x * fastrcp(1.f + __expf(-x)); }
__device__ __forceinline__ float sigmf(float x) { return fastrcp(1.f + __expf(-x)); }

// Load one 16-half bf16 fragment: two contiguous 16B chunks at p and p+16 elems.
__device__ __forceinline__ v16bf load_frag(const bf16_t* p) {
  v8bf a = *(const v8bf*)p;
  v8bf b = *(const v8bf*)(p + 16);
  return __builtin_shufflevector(a, b, 0,1,2,3,4,5,6,7,8,9,10,11,12,13,14,15);
}

// Blocked bf16 activation layout: [P/16][K/32][16(n)][32(k)]
__device__ __forceinline__ size_t bofs(int ptile, int ktile, int K) {
  return ((size_t)ptile * (K >> 5) + ktile) * 512;
}

// ---------------------------------------------------------------------------
// LDS-free WMMA GEMM: out[M][P] = act(A[M][K](bf16,row-major) *
//                                     B(blocked bf16) + bias[M])
// Block 256 thr = 8 waves; block tile 32(M) x 128(P); wave tile 16x32.
// ---------------------------------------------------------------------------
__global__ __launch_bounds__(256) void wmma_gemm_kernel(
    const bf16_t* __restrict__ A, const bf16_t* __restrict__ B,
    int M, int K, int P,
    const float* __restrict__ bias, int act,
    float* __restrict__ outF)
{
  const int t    = threadIdx.x;
  const int lane = t & 31;
  const int wave = t >> 5;
  const int hi   = lane >> 4;
  const int l16  = lane & 15;
  const int mw   = wave & 1;     // M sub-tile 0..1
  const int nw   = wave >> 1;    // N sub-group 0..3 (32 cols)
  const int m0   = blockIdx.y * 32;
  const int n0   = blockIdx.x * 128;
  const int K32  = K >> 5;
  const int tn0  = (n0 >> 4) + nw * 2;     // first 16-col tile of this wave

  v8f acc0, acc1;
#pragma unroll
  for (int i = 0; i < 8; ++i) { acc0[i] = 0.f; acc1[i] = 0.f; }

  const bf16_t* pa = A + (size_t)(m0 + mw * 16 + l16) * K + 8 * hi;
  const bf16_t* pb0 = B + bofs(tn0,     0, K) + (l16 << 5) + (hi << 3);
  const bf16_t* pb1 = B + bofs(tn0 + 1, 0, K) + (l16 << 5) + (hi << 3);

  for (int kt = 0; kt < K32; ++kt) {
    v16bf af  = load_frag(pa + kt * 32);
    v16bf bf0 = load_frag(pb0 + (size_t)kt * 512);
    v16bf bf1 = load_frag(pb1 + (size_t)kt * 512);
    acc0 = __builtin_amdgcn_wmma_f32_16x16x32_bf16(false, af, false, bf0, (short)0, acc0, false, false);
    acc1 = __builtin_amdgcn_wmma_f32_16x16x32_bf16(false, af, false, bf1, (short)0, acc1, false, false);
  }

#pragma unroll
  for (int r = 0; r < 8; ++r) {
    int row = m0 + mw * 16 + r + 8 * hi;
    float bb = bias[row];
    float v0 = acc0[r] + bb;
    float v1 = acc1[r] + bb;
    if (act) { v0 = siluf(v0); v1 = siluf(v1); }
    size_t base = (size_t)row * P + (size_t)(n0 + nw * 32 + l16);
    outF[base]      = v0;
    outF[base + 16] = v1;
  }
}

// ---------------------------------------------------------------------------
// Fused deformable conv: bilinear gather -> blocked B tile in LDS -> WMMA
// K=5184, M=288 (18 subtiles over 8 waves), 16 pixels per block.
// ---------------------------------------------------------------------------
__global__ __launch_bounds__(256) void deform_wmma_kernel(
    const bf16_t* __restrict__ Aw,    // [288][5184]
    const float*  __restrict__ xcat,  // [576][16384]
    const float*  __restrict__ offp,  // [18][16384]
    float*        __restrict__ falign)// [288][16384]
{
  __shared__ int   sAddr[9][16][4];
  __shared__ float sWgt[9][16][4];
  __shared__ __align__(16) bf16_t Bs[16][32];   // [n][k] : k contiguous

  const int t    = threadIdx.x;
  const int lane = t & 31;
  const int wave = t >> 5;
  const int hi   = lane >> 4;
  const int l16  = lane & 15;
  const int pbase = blockIdx.x * 16;

  if (t < 144) {                      // corner table: one (tap, pixel) each
    int k = t >> 4, nn = t & 15;
    int p = pbase + nn;
    int n = p >> 12, hw = p & 4095;
    int y = hw >> 6, x = hw & 63;
    float dy = offp[(size_t)k * NP + p];
    float dx = offp[(size_t)(9 + k) * NP + p];
    float py = (float)(y + (k / 3) - 1) + dy;
    float px = (float)(x + (k % 3) - 1) + dx;
    float fy = floorf(py), fx = floorf(px);
    float wy = py - fy, wx = px - fx;
    int iy0 = (int)fy, ix0 = (int)fx;
    float cw[4] = {(1.f - wy) * (1.f - wx), (1.f - wy) * wx,
                   wy * (1.f - wx),         wy * wx};
#pragma unroll
    for (int i = 0; i < 4; ++i) {
      int cy = iy0 + (i >> 1), cx = ix0 + (i & 1);
      bool valid = (cy >= 0) && (cy < 64) && (cx >= 0) && (cx < 64);
      int cyc = min(max(cy, 0), 63), cxc = min(max(cx, 0), 63);
      sAddr[k][nn][i] = n * 4096 + cyc * 64 + cxc;
      sWgt[k][nn][i]  = valid ? cw[i] : 0.f;
    }
  }
  __syncthreads();

  v8f acc0, acc1, acc2;
#pragma unroll
  for (int i = 0; i < 8; ++i) { acc0[i] = 0.f; acc1[i] = 0.f; acc2[i] = 0.f; }

  const bf16_t* pa = Aw + (size_t)(wave * 16 + l16) * KDFM + 8 * hi;

  for (int kk0 = 0; kk0 < KDFM; kk0 += 32) {
#pragma unroll
    for (int it = 0; it < 2; ++it) {   // sample 32x16 B tile
      int e = t + it * 256;
      int r = e >> 4, nn = e & 15;
      int kk = kk0 + r;
      int c = kk / 9, tap = kk - c * 9;
      const float* xc = xcat + (size_t)c * NP;
      float v = sWgt[tap][nn][0] * xc[sAddr[tap][nn][0]]
              + sWgt[tap][nn][1] * xc[sAddr[tap][nn][1]]
              + sWgt[tap][nn][2] * xc[sAddr[tap][nn][2]]
              + sWgt[tap][nn][3] * xc[sAddr[tap][nn][3]];
      Bs[nn][r] = (bf16_t)v;
    }
    __syncthreads();

    v16bf bfrag = load_frag(&Bs[l16][8 * hi]);
    v16bf a0 = load_frag(pa + kk0);
    acc0 = __builtin_amdgcn_wmma_f32_16x16x32_bf16(false, a0, false, bfrag, (short)0, acc0, false, false);
    v16bf a1 = load_frag(pa + (size_t)128 * KDFM + kk0);   // +8 subtiles
    acc1 = __builtin_amdgcn_wmma_f32_16x16x32_bf16(false, a1, false, bfrag, (short)0, acc1, false, false);
    if (wave < 2) {
      v16bf a2 = load_frag(pa + (size_t)256 * KDFM + kk0); // +16 subtiles
      acc2 = __builtin_amdgcn_wmma_f32_16x16x32_bf16(false, a2, false, bfrag, (short)0, acc2, false, false);
    }
    __syncthreads();
  }

#pragma unroll
  for (int r = 0; r < 8; ++r) {
    int rr = r + 8 * hi;
    falign[(size_t)(wave * 16 + rr) * NP + pbase + l16]         = siluf(acc0[r]);
    falign[(size_t)((wave + 8) * 16 + rr) * NP + pbase + l16]   = siluf(acc1[r]);
  }
  if (wave < 2) {
#pragma unroll
    for (int r = 0; r < 8; ++r) {
      int rr = r + 8 * hi;
      falign[(size_t)((wave + 16) * 16 + rr) * NP + pbase + l16] = siluf(acc2[r]);
    }
  }
}

// ---------------------------------------------------------------------------
// 3x3 offset-prediction conv (576 -> 18, padded to 32) via shifted-tap WMMA
// ---------------------------------------------------------------------------
__global__ __launch_bounds__(256) void offpred_wmma_kernel(
    const bf16_t* __restrict__ Ap,    // [32][5184], rows 18..31 zero
    const float*  __restrict__ xcat,  // [576][16384]
    const float*  __restrict__ bias18,
    float*        __restrict__ offp)  // [18][16384]
{
  __shared__ int   sAddr[9][16];
  __shared__ float sWgt[9][16];
  __shared__ __align__(16) bf16_t Bs[16][32];

  const int t    = threadIdx.x;
  const int lane = t & 31;
  const int wave = t >> 5;
  const int hi   = lane >> 4;
  const int l16  = lane & 15;
  const int pbase = blockIdx.x * 16;

  if (t < 144) {
    int k = t >> 4, nn = t & 15;
    int p = pbase + nn;
    int n = p >> 12, hw = p & 4095;
    int y = hw >> 6, x = hw & 63;
    int yy = y + (k / 3) - 1, xx = x + (k % 3) - 1;
    bool valid = (yy >= 0) && (yy < 64) && (xx >= 0) && (xx < 64);
    int yc = min(max(yy, 0), 63), xc = min(max(xx, 0), 63);
    sAddr[k][nn] = n * 4096 + yc * 64 + xc;
    sWgt[k][nn]  = valid ? 1.f : 0.f;
  }
  __syncthreads();

  v8f acc;
#pragma unroll
  for (int i = 0; i < 8; ++i) acc[i] = 0.f;

  const bf16_t* pa = Ap + (size_t)(wave * 16 + l16) * KDFM + 8 * hi;

  for (int kk0 = 0; kk0 < KDFM; kk0 += 32) {
#pragma unroll
    for (int it = 0; it < 2; ++it) {
      int e = t + it * 256;
      int r = e >> 4, nn = e & 15;
      int kk = kk0 + r;
      int c = kk / 9, tap = kk - c * 9;
      float v = sWgt[tap][nn] * xcat[(size_t)c * NP + sAddr[tap][nn]];
      Bs[nn][r] = (bf16_t)v;
    }
    __syncthreads();
    if (wave < 2) {
      v16bf bfrag = load_frag(&Bs[l16][8 * hi]);
      v16bf afrag = load_frag(pa + kk0);
      acc = __builtin_amdgcn_wmma_f32_16x16x32_bf16(false, afrag, false, bfrag, (short)0, acc, false, false);
    }
    __syncthreads();
  }
  if (wave < 2) {
#pragma unroll
    for (int r = 0; r < 8; ++r) {
      int row = wave * 16 + r + 8 * hi;
      if (row < 18)
        offp[(size_t)row * NP + pbase + l16] = acc[r] + bias18[row];
    }
  }
}

// ------------------------- small helper kernels ----------------------------
__global__ void zero_f32_kernel(float* p, int n) {
  int t = blockIdx.x * blockDim.x + threadIdx.x;
  if (t < n) p[t] = 0.f;
}

__global__ void fold_bn_kernel(const float* g, const float* b, const float* m,
                               const float* v, float* scale, float* bias, int n) {
  int t = blockIdx.x * blockDim.x + threadIdx.x;
  if (t >= n) return;
  float s = g[t] * rsqrtf(v[t] + EPSBN);
  scale[t] = s;
  bias[t]  = b[t] - m[t] * s;
}

__global__ void fold_convert_kernel(bf16_t* dst, const float* src,
                                    const float* scale, int dstM, int srcM, int K) {
  size_t t = (size_t)blockIdx.x * blockDim.x + threadIdx.x;
  if (t >= (size_t)dstM * K) return;
  int m = (int)(t / K);
  float v = 0.f;
  if (m < srcM) {
    v = src[t];
    if (scale) v *= scale[m];
  }
  dst[t] = (bf16_t)v;
}

__global__ void prep_up_kernel(bf16_t* dst, const float* up_w, const float* s_up) {
  size_t t = (size_t)blockIdx.x * blockDim.x + threadIdx.x;
  if (t >= (size_t)4 * CIN * CIN) return;
  int dk = (int)(t / (CIN * CIN));
  int r  = (int)(t % (CIN * CIN));
  int o = r / CIN, c = r % CIN;
  int d = dk >> 1, kq = dk & 1;
  dst[t] = (bf16_t)(up_w[(((size_t)c * CIN + o) * 2 + d) * 2 + kq] * s_up[o]);
}

// NCHW fp32 -> blocked bf16 (32-channel gather; coalesced reads and writes)
__global__ void nchw_to_blocked_kernel(bf16_t* dst, const float* src, int HW) {
  int t = blockIdx.x * blockDim.x + threadIdx.x;
  int Ptot = 4 * HW;
  if (t >= (CIN / 32) * Ptot) return;
  int cb = t / Ptot, p = t % Ptot;
  int n = p / HW, hw = p % HW;
  const float* s = src + ((size_t)n * CIN + cb * 32) * HW + hw;
  v8bf ch[4];
#pragma unroll
  for (int j = 0; j < 32; ++j) ch[j >> 3][j & 7] = (bf16_t)s[(size_t)j * HW];
  bf16_t* d = dst + bofs(p >> 4, cb, CIN) + ((p & 15) << 5);
#pragma unroll
  for (int i = 0; i < 4; ++i) *(v8bf*)(d + i * 8) = ch[i];
}

// y4 [dk][c][ps] -> blocked bf16 feat_up_a at full res
__global__ void upsample_kernel(const float* y4, bf16_t* dst) {
  int t = blockIdx.x * blockDim.x + threadIdx.x;
  if (t >= (CIN / 32) * NP) return;
  int cb = t / NP, p = t % NP;
  int n = p >> 12, hw = p & 4095;
  int y = hw >> 6, x = hw & 63;
  int d = y & 1, kq = x & 1;
  int dk = d * 2 + kq;
  int ps = n * 1024 + (y >> 1) * 32 + (x >> 1);
  const float* s = y4 + ((size_t)dk * CIN + cb * 32) * NPS + ps;
  v8bf ch[4];
#pragma unroll
  for (int j = 0; j < 32; ++j) ch[j >> 3][j & 7] = (bf16_t)s[(size_t)j * NPS];
  bf16_t* dp = dst + bofs(p >> 4, cb, CIN) + ((p & 15) << 5);
#pragma unroll
  for (int i = 0; i < 4; ++i) *(v8bf*)(dp + i * 8) = ch[i];
}

// fp32 [rows][NP] -> blocked bf16 (with scale, row offset into 576-row dest)
__global__ void cp_to_blocked_kernel(bf16_t* dst, const float* src,
                                     float scale, int rows, int rowOff) {
  int t = blockIdx.x * blockDim.x + threadIdx.x;
  if (t >= (rows / 32) * NP) return;
  int cb = t / NP, p = t % NP;
  const float* s = src + (size_t)cb * 32 * NP + p;
  v8bf ch[4];
#pragma unroll
  for (int j = 0; j < 32; ++j) ch[j >> 3][j & 7] = (bf16_t)(s[(size_t)j * NP] * scale);
  bf16_t* d = dst + bofs(p >> 4, (rowOff >> 5) + cb, CIN) + ((p & 15) << 5);
#pragma unroll
  for (int i = 0; i < 4; ++i) *(v8bf*)(d + i * 8) = ch[i];
}

// x2 = g2 (silu(bn(conv)) already) + feat_l * atten  -> blocked bf16
__global__ void x2_kernel(const float* g2, const float* feat_l,
                          const float* atten, bf16_t* dst) {
  int t = blockIdx.x * blockDim.x + threadIdx.x;
  if (t >= (CIN / 32) * NP) return;
  int cb = t / NP, p = t % NP;
  int n = p >> 12, hw = p & 4095;
  const float* s1 = g2 + (size_t)cb * 32 * NP + p;
  const float* s2 = feat_l + ((size_t)n * CIN + cb * 32) * 4096 + hw;
  const float* at = atten + n * CIN + cb * 32;
  v8bf ch[4];
#pragma unroll
  for (int j = 0; j < 32; ++j)
    ch[j >> 3][j & 7] = (bf16_t)(s1[(size_t)j * NP] + s2[(size_t)j * 4096] * at[j]);
  bf16_t* d = dst + bofs(p >> 4, cb, CIN) + ((p & 15) << 5);
#pragma unroll
  for (int i = 0; i < 4; ++i) *(v8bf*)(d + i * 8) = ch[i];
}

__global__ void pool_kernel(const float* feat_l, float* pooled) {
  int wave = threadIdx.x >> 5, lane = threadIdx.x & 31;
  int g = blockIdx.x * 8 + wave;                // 0..2303 = n*576+c
  const float* src = feat_l + (size_t)g * 4096;
  float s = 0.f;
  for (int i = lane; i < 4096; i += 32) s += src[i];
  for (int m = 16; m >= 1; m >>= 1) s += __shfl_xor(s, m, 32);
  if (lane == 0) pooled[g] = s * (1.f / 4096.f);
}

__global__ void atten_kernel(const float* aw, const float* pooled,
                             const float* s_in, const float* b_in, float* atten) {
  int t = blockIdx.x * blockDim.x + threadIdx.x;
  if (t >= 4 * CIN) return;
  int n = t / CIN, o = t % CIN;
  float s = 0.f;
  for (int c = 0; c < CIN; ++c) s += aw[(size_t)o * CIN + c] * pooled[n * CIN + c];
  float x = s * s_in[o] + b_in[o];   // folded BN
  atten[t] = sigmf(siluf(x));
}

__global__ __launch_bounds__(256) void gate_blend_kernel(
    const float* falign, const float* farm,
    const float* gate_w, const float* gate_b, float* out) {
  int wave = threadIdx.x >> 5, lane = threadIdx.x & 31;
  int p = blockIdx.x * 8 + wave;                // 0..16383
  float s = 0.f;
#pragma unroll
  for (int j = 0; j < 9; ++j)
    s += gate_w[lane + j * 32] * falign[(size_t)(lane + j * 32) * NP + p];
#pragma unroll
  for (int j = 0; j < 9; ++j)
    s += gate_w[288 + lane + j * 32] * farm[(size_t)(lane + j * 32) * NP + p];
  for (int m = 16; m >= 1; m >>= 1) s += __shfl_xor(s, m, 32);
  float g = tanhf(s + gate_b[0]);
  int n = p >> 12, hw = p & 4095;
#pragma unroll
  for (int j = 0; j < 9; ++j) {
    int o = lane + j * 32;
    float a = falign[(size_t)o * NP + p];
    float r = farm[(size_t)o * NP + p];
    out[((size_t)n * COUT + o) * 4096 + hw] = g * a + (1.f - g) * r;
  }
}

// ---------------------------------------------------------------------------
extern "C" void kernel_launch(void* const* d_in, const int* in_sizes, int n_in,
                              void* d_out, int out_size, void* d_ws, size_t ws_size,
                              hipStream_t stream) {
  const float* feat_l     = (const float*)d_in[0];
  const float* feat_s     = (const float*)d_in[1];
  const float* ise_atten_w= (const float*)d_in[2];
  const float* ise_lat_w  = (const float*)d_in[3];
  const float* ise_conv_w = (const float*)d_in[4];
  const float* bnin_g = (const float*)d_in[5],  *bnin_b = (const float*)d_in[6];
  const float* bnin_m = (const float*)d_in[7],  *bnin_v = (const float*)d_in[8];
  const float* bnout_g= (const float*)d_in[9],  *bnout_b= (const float*)d_in[10];
  const float* bnout_m= (const float*)d_in[11], *bnout_v= (const float*)d_in[12];
  const float* up_w   = (const float*)d_in[13];
  const float* bnup_g = (const float*)d_in[14], *bnup_b = (const float*)d_in[15];
  const float* bnup_m = (const float*)d_in[16], *bnup_v = (const float*)d_in[17];
  const float* lat_w  = (const float*)d_in[18];
  const float* off_w  = (const float*)d_in[19];
  const float* bn_g   = (const float*)d_in[20], *bn_b   = (const float*)d_in[21];
  const float* bn_m   = (const float*)d_in[22], *bn_v   = (const float*)d_in[23];
  const float* dfm_p_w= (const float*)d_in[24];
  const float* dfm_p_b= (const float*)d_in[25];
  const float* dfm_w  = (const float*)d_in[26];
  const float* gate_w = (const float*)d_in[27];
  const float* gate_b = (const float*)d_in[28];
  float* out = (float*)d_out;

  size_t off = 0;
  auto alloc = [&](size_t bytes) -> void* {
    void* p = (char*)d_ws + off;
    off += (bytes + 255) & ~(size_t)255;
    return p;
  };
  float* s_up  = (float*)alloc(CIN * 4);   float* b_up  = (float*)alloc(CIN * 4);
  float* s_in  = (float*)alloc(CIN * 4);   float* b_in  = (float*)alloc(CIN * 4);
  float* s_out = (float*)alloc(COUT * 4);  float* b_out = (float*)alloc(COUT * 4);
  float* s_off = (float*)alloc(COUT * 4);  float* b_off = (float*)alloc(COUT * 4);
  float* b_zero= (float*)alloc(CIN * 4);
  float* pooled = (float*)alloc(4 * CIN * 4);
  float* atten  = (float*)alloc(4 * CIN * 4);
  bf16_t* wq_up     = (bf16_t*)alloc((size_t)4 * CIN * CIN * 2);
  bf16_t* wq_lat    = (bf16_t*)alloc((size_t)COUT * CIN * 2);
  bf16_t* wq_iselat = (bf16_t*)alloc((size_t)CIN * CIN * 2);
  bf16_t* wq_iseconv= (bf16_t*)alloc((size_t)COUT * CIN * 2);
  bf16_t* wq_off    = (bf16_t*)alloc((size_t)COUT * CIN * 2);
  bf16_t* wq_dfm    = (bf16_t*)alloc((size_t)COUT * KDFM * 2);
  bf16_t* wq_dfmp   = (bf16_t*)alloc((size_t)32 * KDFM * 2);
  bf16_t* xs_bf     = (bf16_t*)alloc((size_t)CIN * NPS * 2);
  bf16_t* xl_bf     = (bf16_t*)alloc((size_t)CIN * NP * 2);
  float*  y4        = (float*)alloc((size_t)4 * CIN * NPS * 4);  // reused as g2raw
  bf16_t* feat_up_a = (bf16_t*)alloc((size_t)CIN * NP * 2);
  bf16_t* concat_a  = (bf16_t*)alloc((size_t)CIN * NP * 2);
  bf16_t* x2_bf     = (bf16_t*)alloc((size_t)CIN * NP * 2);
  float*  x_cat     = (float*)alloc((size_t)CIN * NP * 4);
  float*  feat_arm  = (float*)alloc((size_t)COUT * NP * 4);
  float*  off_pred  = (float*)alloc((size_t)18 * NP * 4);
  float*  feat_align= (float*)alloc((size_t)COUT * NP * 4);
  float*  g2raw = y4;

  auto cdiv = [](size_t a, size_t b) { return (unsigned)((a + b - 1) / b); };

  // --- fold BN params, zero bias ---
  zero_f32_kernel<<<cdiv(CIN, 256), 256, 0, stream>>>(b_zero, CIN);
  fold_bn_kernel<<<cdiv(CIN, 256), 256, 0, stream>>>(bnup_g, bnup_b, bnup_m, bnup_v, s_up, b_up, CIN);
  fold_bn_kernel<<<cdiv(CIN, 256), 256, 0, stream>>>(bnin_g, bnin_b, bnin_m, bnin_v, s_in, b_in, CIN);
  fold_bn_kernel<<<cdiv(COUT, 256), 256, 0, stream>>>(bnout_g, bnout_b, bnout_m, bnout_v, s_out, b_out, COUT);
  fold_bn_kernel<<<cdiv(COUT, 256), 256, 0, stream>>>(bn_g, bn_b, bn_m, bn_v, s_off, b_off, COUT);

  // --- weight conversion / folding ---
  prep_up_kernel<<<cdiv((size_t)4 * CIN * CIN, 256), 256, 0, stream>>>(wq_up, up_w, s_up);
  fold_convert_kernel<<<cdiv((size_t)COUT * CIN, 256), 256, 0, stream>>>(wq_lat, lat_w, nullptr, COUT, COUT, CIN);
  fold_convert_kernel<<<cdiv((size_t)CIN * CIN, 256), 256, 0, stream>>>(wq_iselat, ise_lat_w, s_in, CIN, CIN, CIN);
  fold_convert_kernel<<<cdiv((size_t)COUT * CIN, 256), 256, 0, stream>>>(wq_iseconv, ise_conv_w, s_out, COUT, COUT, CIN);
  fold_convert_kernel<<<cdiv((size_t)COUT * CIN, 256), 256, 0, stream>>>(wq_off, off_w, s_off, COUT, COUT, CIN);
  fold_convert_kernel<<<cdiv((size_t)COUT * KDFM, 256), 256, 0, stream>>>(wq_dfm, dfm_w, nullptr, COUT, COUT, KDFM);
  fold_convert_kernel<<<cdiv((size_t)32 * KDFM, 256), 256, 0, stream>>>(wq_dfmp, dfm_p_w, nullptr, 32, 18, KDFM);

  // --- inputs to blocked bf16 ---
  nchw_to_blocked_kernel<<<cdiv((size_t)(CIN / 32) * NPS, 256), 256, 0, stream>>>(xs_bf, feat_s, 1024);
  nchw_to_blocked_kernel<<<cdiv((size_t)(CIN / 32) * NP, 256), 256, 0, stream>>>(xl_bf, feat_l, 4096);

  // --- ConvTranspose (4 GEMMs, BN+SiLU fused) + scatter upsample ---
  for (int dk = 0; dk < 4; ++dk)
    wmma_gemm_kernel<<<dim3(NPS / 128, CIN / 32), 256, 0, stream>>>(
        wq_up + (size_t)dk * CIN * CIN, xs_bf, CIN, CIN, NPS,
        b_up, 1, y4 + (size_t)dk * CIN * NPS);
  upsample_kernel<<<cdiv((size_t)(CIN / 32) * NP, 256), 256, 0, stream>>>(y4, feat_up_a);

  // --- lat conv (SiLU): fp32 -> x_cat rows [0,288); bf16*2 -> concat rows [288,576)
  wmma_gemm_kernel<<<dim3(NP / 128, COUT / 32), 256, 0, stream>>>(
      wq_lat, feat_up_a, COUT, CIN, NP, b_zero, 1, x_cat);
  cp_to_blocked_kernel<<<cdiv((size_t)(COUT / 32) * NP, 256), 256, 0, stream>>>(
      concat_a, x_cat, 2.f, COUT, COUT);

  // --- ISE path ---
  pool_kernel<<<4 * CIN / 8, 256, 0, stream>>>(feat_l, pooled);
  atten_kernel<<<cdiv(4 * CIN, 256), 256, 0, stream>>>(ise_atten_w, pooled, s_in, b_in, atten);
  wmma_gemm_kernel<<<dim3(NP / 128, CIN / 32), 256, 0, stream>>>(
      wq_iselat, xl_bf, CIN, CIN, NP, b_in, 1, g2raw);
  x2_kernel<<<cdiv((size_t)(CIN / 32) * NP, 256), 256, 0, stream>>>(g2raw, feat_l, atten, x2_bf);
  wmma_gemm_kernel<<<dim3(NP / 128, COUT / 32), 256, 0, stream>>>(
      wq_iseconv, x2_bf, COUT, CIN, NP, b_out, 1, feat_arm);
  cp_to_blocked_kernel<<<cdiv((size_t)(COUT / 32) * NP, 256), 256, 0, stream>>>(
      concat_a, feat_arm, 1.f, COUT, 0);

  // --- offset conv (BN, no act): fp32 -> x_cat rows [288,576) ---
  wmma_gemm_kernel<<<dim3(NP / 128, COUT / 32), 256, 0, stream>>>(
      wq_off, concat_a, COUT, CIN, NP, b_off, 0, x_cat + (size_t)COUT * NP);

  // --- deformable conv ---
  offpred_wmma_kernel<<<NP / 16, 256, 0, stream>>>(wq_dfmp, x_cat, dfm_p_b, off_pred);
  deform_wmma_kernel<<<NP / 16, 256, 0, stream>>>(wq_dfm, x_cat, off_pred, feat_align);

  // --- gate + blend ---
  gate_blend_kernel<<<NP / 8, 256, 0, stream>>>(feat_align, feat_arm, gate_w, gate_b, out);

  (void)in_sizes; (void)n_in; (void)out_size; (void)ws_size;
}
